// FeedForward_7559142441191
// MI455X (gfx1250) — matrix operand order repaired and checked by reading, not compiled, not run on hardware
//
#include <hip/hip_runtime.h>
#include <hip/hip_bf16.h>

// ---------------------------------------------------------------------------
// MoE FFN for MI455X (gfx1250, wave32, WMMA + Tensor Data Mover).
// Top-K-only grouped GEMM pipeline, bf16 WMMA 16x16x32, f32 accumulate.
// ---------------------------------------------------------------------------

typedef __attribute__((ext_vector_type(16))) __bf16 v16bf;
typedef __attribute__((ext_vector_type(8)))  float  v8f;

#define TOKENS   8192      // B*S
#define DIM      1024
#define HID      4096
#define NEXP     8
#define TOPK     4
#define TILE_M   128
#define TILE_N   128
#define TILE_K   32
#define LDSS     48        // padded LDS row stride (bf16 elems): 96B, 16B-aligned
#define MAX_ROWS 33792     // 32768 assignments + 8*128 tile padding

#if __has_builtin(__builtin_amdgcn_tensor_load_to_lds)
#define HAVE_TDM 1
#else
#define HAVE_TDM 0
#endif

union FragU { int4 q[2]; v16bf v; };

__device__ __forceinline__ v16bf ldsFragA(const __bf16* base, int lane) {
    // A 16x32 bf16: lanes 0-15 -> row=lane, K {0..7,16..23}; lanes 16-31 -> K {8..15,24..31}
    int half = lane >> 4, lr = lane & 15;
    FragU f;
    f.q[0] = *(const int4*)(base + lr * LDSS + half * 8);
    f.q[1] = *(const int4*)(base + lr * LDSS + half * 8 + 16);
    return f.v;
}

__device__ __forceinline__ v16bf ldsFragB(const __bf16* base, int lane) {
    // B 32x16 bf16 (N-major in LDS): lanes 0-15 -> col=lane, K {0..7,16..23}; lanes 16-31 -> K {8..15,24..31}
    int half = lane >> 4, lr = lane & 15;
    FragU f;
    f.q[0] = *(const int4*)(base + lr * LDSS + half * 8);
    f.q[1] = *(const int4*)(base + lr * LDSS + half * 8 + 16);
    return f.v;
}

__device__ __forceinline__ float gelu_tanh(float x) {
    float x3 = x * x * x;
    float t  = tanhf(0.7978845608028654f * (x + 0.044715f * x3));
    return 0.5f * x * (1.0f + t);
}

// ---------------------------------------------------------------------------
// Tensor Data Mover: 2D tile load (tile 32 x 128 rows, 2B elems) into LDS with
// per-row padding 64B->96B so LDS layout matches LDSS. D# per ISA 8.3/8.4.
// Flat LDS address low 32 bits == LDS byte offset (aperture in high word).
// This toolchain exposes the 6-arg builtin (clang-23 / therock-10.0 form):
//   (uint32x4 g0, int32x8 g1, int32x4 g2, int32x4 g3, int32x8, i32 cpol)
// ---------------------------------------------------------------------------
#if HAVE_TDM
typedef __attribute__((ext_vector_type(4))) unsigned int tdm_g0_t;
typedef __attribute__((ext_vector_type(8))) int          tdm_g1_t;
typedef __attribute__((ext_vector_type(4))) int          tdm_g2_t;

__device__ __forceinline__ void tdm_load_tile(const void* lds_dst, const void* gsrc,
                                              unsigned rowStrideElems) {
    unsigned long long ga = (unsigned long long)(size_t)gsrc;
    tdm_g0_t g0;
    g0.x = 1u;                                            // count=1, user flags 0
    g0.y = (unsigned)(size_t)lds_dst;                     // lds_addr (byte offset)
    g0.z = (unsigned)ga;                                  // global_addr[31:0]
    g0.w = (unsigned)((ga >> 32) & 0x01FFFFFFull) | (2u << 30);   // addr[56:32], type=2
    tdm_g1_t g1;
    g1[0] = (int)((1u << 16)      // data_size = 2 bytes
                | (1u << 20)      // pad_enable
                | (3u << 22)      // pad_interval: every 64B
                | (7u << 25));    // pad_amount: 8 dwords = 32B  -> 96B LDS row stride
    g1[1] = (int)((0xFFFFFu & 0xFFFFu) << 16);            // tensor_dim0[15:0] (large)
    g1[2] = (int)(((0xFFFFFu >> 16) & 0xFFFFu)            // tensor_dim0[31:16]
                | ((0xFFFFFu & 0xFFFFu) << 16));          // tensor_dim1[15:0] (large)
    g1[3] = (int)(((0xFFFFFu >> 16) & 0xFFFFu)            // tensor_dim1[31:16]
                | ((unsigned)TILE_K << 16));              // tile_dim0 = 32 elems
    g1[4] = (int)(unsigned)TILE_M;                        // tile_dim1 = 128 rows
    g1[5] = (int)rowStrideElems;                          // tensor_dim0_stride[31:0]
    g1[6] = 0;
    g1[7] = 0;
    __builtin_amdgcn_tensor_load_to_lds(g0, g1, (tdm_g2_t)0, (tdm_g2_t)0,
                                        (tdm_g1_t)0, 0);
}
#endif

__device__ __forceinline__ void tdm_wait() {
#if __has_builtin(__builtin_amdgcn_s_wait_tensorcnt)
    __builtin_amdgcn_s_wait_tensorcnt(0);
#else
    asm volatile("s_wait_tensorcnt 0x0" ::: "memory");
#endif
}

// ---------------------------------------------------------------------------
// fp32 -> bf16 conversion of activations
// ---------------------------------------------------------------------------
__global__ void convert_x_bf16(const float* __restrict__ x, __bf16* __restrict__ xb) {
    int i = (blockIdx.x * blockDim.x + threadIdx.x) * 4;
    float4 v = *(const float4*)(x + i);
    union { __bf16 h[4]; short4 s; } u;
    u.h[0] = (__bf16)v.x; u.h[1] = (__bf16)v.y; u.h[2] = (__bf16)v.z; u.h[3] = (__bf16)v.w;
    *(short4*)(xb + i) = u.s;
}

// per-expert [R,C] fp32 -> [C,R] bf16 (N-major weights, contiguous K rows)
__global__ void transpose_to_bf16(const float* __restrict__ src, __bf16* __restrict__ dst,
                                  int R, int C) {
    __shared__ float tile[32][33];
    int z = blockIdx.z;
    const float* s = src + (size_t)z * R * C;
    __bf16*      d = dst + (size_t)z * R * C;
    int c0 = blockIdx.x * 32, r0 = blockIdx.y * 32;
    int tx = threadIdx.x & 31, ty = threadIdx.x >> 5;   // 32x8
#pragma unroll
    for (int i = 0; i < 4; ++i)
        tile[ty + i * 8][tx] = s[(size_t)(r0 + ty + i * 8) * C + c0 + tx];
    __syncthreads();
#pragma unroll
    for (int i = 0; i < 4; ++i)
        d[(size_t)(c0 + ty + i * 8) * R + r0 + tx] = (__bf16)tile[tx][ty + i * 8];
}

// ---------------------------------------------------------------------------
// Router: logits, softmax-exp, exact top-4 (lowest index wins ties),
// renormalized weights, per-expert counts. One wave per token.
// ---------------------------------------------------------------------------
__global__ __launch_bounds__(256)
void router_topk_kernel(const float* __restrict__ x, const float* __restrict__ rw,
                        float* __restrict__ wts, int* __restrict__ idx,
                        int* __restrict__ counts) {
    int lane  = threadIdx.x & 31;
    int wave  = threadIdx.x >> 5;
    int token = blockIdx.x * 8 + wave;
    float acc[NEXP];
#pragma unroll
    for (int e = 0; e < NEXP; ++e) acc[e] = 0.f;
    const float* xr = x + (size_t)token * DIM;
    for (int j = 0; j < DIM / 32; ++j) {
        int dd = lane + 32 * j;
        float xv = xr[dd];
        const float* r = rw + (size_t)dd * NEXP;
#pragma unroll
        for (int e = 0; e < NEXP; ++e) acc[e] += xv * r[e];
    }
#pragma unroll
    for (int off = 16; off > 0; off >>= 1)
#pragma unroll
        for (int e = 0; e < NEXP; ++e) acc[e] += __shfl_xor(acc[e], off, 32);
    if (lane == 0) {
        float mx = acc[0];
#pragma unroll
        for (int e = 1; e < NEXP; ++e) mx = fmaxf(mx, acc[e]);
        float p[NEXP];
#pragma unroll
        for (int e = 0; e < NEXP; ++e) p[e] = __expf(acc[e] - mx);
        bool used[NEXP] = {};
        float s = 0.f;
        int   sel[TOPK];
        float sw[TOPK];
#pragma unroll
        for (int k = 0; k < TOPK; ++k) {
            float best = -1.f; int bi = 0;
#pragma unroll
            for (int e = 0; e < NEXP; ++e)
                if (!used[e] && p[e] > best) { best = p[e]; bi = e; }
            used[bi] = true; sel[k] = bi; sw[k] = best; s += best;
        }
        float inv = 1.f / s;
#pragma unroll
        for (int k = 0; k < TOPK; ++k) {
            wts[token * TOPK + k] = sw[k] * inv;
            idx[token * TOPK + k] = sel[k];
            atomicAdd(&counts[sel[k]], 1);
        }
    }
}

// 8-element exclusive scan of tile-padded counts
__global__ void scan_kernel(const int* __restrict__ counts, int* __restrict__ offsets) {
    if (threadIdx.x == 0 && blockIdx.x == 0) {
        int o = 0;
        for (int e = 0; e < NEXP; ++e) {
            offsets[e] = o;
            o += ((counts[e] + TILE_M - 1) / TILE_M) * TILE_M;
        }
    }
}

// Build per-expert token lists + per-token row map (output is permutation-invariant)
__global__ void scatter_kernel(const int* __restrict__ idx, const int* __restrict__ offsets,
                               int* __restrict__ fill, int* __restrict__ tokList,
                               int* __restrict__ rowOf) {
    int t = blockIdx.x * blockDim.x + threadIdx.x;
#pragma unroll
    for (int k = 0; k < TOPK; ++k) {
        int e    = idx[t * TOPK + k];
        int slot = atomicAdd(&fill[e], 1);
        int row  = offsets[e] + slot;
        tokList[row]        = t;
        rowOf[t * TOPK + k] = row;
    }
}

// ---------------------------------------------------------------------------
// GEMM1: Hbuf[row, :] = gelu( xb[tok(row), :] @ upT[e].T )
// 128x128 tile, K-step 32, double-buffered LDS. A staged by threads (gather),
// B tiles moved by TDM. C restaged through LDS for coalesced b128 stores.
// ---------------------------------------------------------------------------
__global__ __launch_bounds__(256)
void gemm_up_kernel(const __bf16* __restrict__ xb, const __bf16* __restrict__ upT,
                    const int* __restrict__ tokList, const int* __restrict__ counts,
                    const int* __restrict__ offsets, __bf16* __restrict__ Hbuf) {
    int e      = blockIdx.z;
    int count  = counts[e];
    int mTiles = (count + TILE_M - 1) / TILE_M;
    if ((int)blockIdx.y >= mTiles) return;
    int rowBase = offsets[e] + blockIdx.y * TILE_M;
    int rowEnd  = offsets[e] + count;
    int nBase   = blockIdx.x * TILE_N;
    const __bf16* Bw = upT + (size_t)e * HID * DIM;

    __shared__ __align__(16) char smem[49152];
    __shared__ int tok[TILE_M];
    __bf16* As = (__bf16*)smem;                 // [2][TILE_M*LDSS]
    __bf16* Bs = (__bf16*)(smem + 24576);       // [2][TILE_M*LDSS]

    int tid = threadIdx.x, lane = tid & 31, wave = tid >> 5;
    int wm = wave >> 1, wn = wave & 1;

    if (tid < TILE_M) {
        int gRow = rowBase + tid;
        tok[tid] = (gRow < rowEnd) ? tokList[gRow] : 0;
    }
    __syncthreads();

    v8f acc[2][4] = {};

    auto stageA = [&](int k0, int buf) {
#pragma unroll
        for (int it = 0; it < 2; ++it) {
            int i = tid + it * 256;              // 512 x 16B chunks
            int row = i >> 2, seg = i & 3;
            *(int4*)(As + buf * TILE_M * LDSS + row * LDSS + seg * 8) =
                *(const int4*)(xb + (size_t)tok[row] * DIM + k0 + seg * 8);
        }
    };
#if !HAVE_TDM
    auto stageB = [&](int k0, int buf) {
#pragma unroll
        for (int it = 0; it < 2; ++it) {
            int i = tid + it * 256;
            int row = i >> 2, seg = i & 3;
            *(int4*)(Bs + buf * TILE_M * LDSS + row * LDSS + seg * 8) =
                *(const int4*)(Bw + (size_t)(nBase + row) * DIM + k0 + seg * 8);
        }
    };
#endif

    stageA(0, 0);
#if HAVE_TDM
    if (wave == 0) {
        tdm_load_tile(Bs, Bw + (size_t)nBase * DIM, DIM);
        tdm_wait();
    }
#else
    stageB(0, 0);
#endif
    __syncthreads();

    const int ksteps = DIM / TILE_K;   // 32
    for (int kt = 0; kt < ksteps; ++kt) {
        int buf = kt & 1;
        if (kt + 1 < ksteps) {
            int k0 = (kt + 1) * TILE_K;
            stageA(k0, buf ^ 1);
#if HAVE_TDM
            if (wave == 0)
                tdm_load_tile(Bs + (buf ^ 1) * TILE_M * LDSS,
                              Bw + (size_t)nBase * DIM + k0, DIM);
#else
            stageB(k0, buf ^ 1);
#endif
        }
        v16bf af[2], bfg[4];
#pragma unroll
        for (int i = 0; i < 2; ++i)
            af[i] = ldsFragA(As + buf * TILE_M * LDSS + (wm * 32 + i * 16) * LDSS, lane);
#pragma unroll
        for (int j = 0; j < 4; ++j)
            bfg[j] = ldsFragB(Bs + buf * TILE_M * LDSS + (wn * 64 + j * 16) * LDSS, lane);
#pragma unroll
        for (int i = 0; i < 2; ++i)
#pragma unroll
            for (int j = 0; j < 4; ++j)
                acc[i][j] = __builtin_amdgcn_wmma_f32_16x16x32_bf16(
                    false, af[i], false, bfg[j], (short)0, acc[i][j], false, false);
#if HAVE_TDM
        if (wave == 0 && kt + 1 < ksteps) tdm_wait();
#endif
        __syncthreads();
    }

    // Epilogue: gelu -> bf16, restage in LDS (padded stride), coalesced b128 stores.
    const int CSTR = 136;                        // 272B rows, 16B-aligned, bank-friendly
    __bf16* Cs = (__bf16*)smem;                  // 128*136*2 = 34816B <= 48KB
    int half = lane >> 4, ln = lane & 15;
#pragma unroll
    for (int i = 0; i < 2; ++i)
#pragma unroll
        for (int j = 0; j < 4; ++j)
#pragma unroll
            for (int r = 0; r < 8; ++r)
                Cs[(wm * 32 + i * 16 + half * 8 + r) * CSTR + wn * 64 + j * 16 + ln] =
                    (__bf16)gelu_tanh(acc[i][j][r]);
    __syncthreads();
#pragma unroll
    for (int it = 0; it < 8; ++it) {
        int c = tid + it * 256;                  // 2048 x 16B chunks
        int r = c >> 4, seg = c & 15;
        *(int4*)(Hbuf + (size_t)(rowBase + r) * HID + nBase + seg * 8) =
            *(const int4*)(Cs + r * CSTR + seg * 8);
    }
}

// ---------------------------------------------------------------------------
// GEMM2: Gout[row, :] = Hbuf[row, :] @ downT[e].T (fp32 out).
// Both A and B tiles are dense 2D tiles -> both moved by TDM.
// ---------------------------------------------------------------------------
__global__ __launch_bounds__(256)
void gemm_down_kernel(const __bf16* __restrict__ Hbuf, const __bf16* __restrict__ downT,
                      const int* __restrict__ counts, const int* __restrict__ offsets,
                      float* __restrict__ Gout) {
    int e      = blockIdx.z;
    int count  = counts[e];
    int mTiles = (count + TILE_M - 1) / TILE_M;
    if ((int)blockIdx.y >= mTiles) return;
    int rowBase = offsets[e] + blockIdx.y * TILE_M;
    int nBase   = blockIdx.x * TILE_N;
    const __bf16* Bw = downT + (size_t)e * DIM * HID;
    const __bf16* Aw = Hbuf + (size_t)rowBase * HID;

    __shared__ __align__(16) char smem[49152];
    __bf16* As = (__bf16*)smem;
    __bf16* Bs = (__bf16*)(smem + 24576);

    int tid = threadIdx.x, lane = tid & 31, wave = tid >> 5;
    int wm = wave >> 1, wn = wave & 1;

    v8f acc[2][4] = {};

#if !HAVE_TDM
    auto stage = [&](int k0, int buf) {
#pragma unroll
        for (int it = 0; it < 2; ++it) {
            int i = tid + it * 256;
            int row = i >> 2, seg = i & 3;
            *(int4*)(As + buf * TILE_M * LDSS + row * LDSS + seg * 8) =
                *(const int4*)(Aw + (size_t)row * HID + k0 + seg * 8);
            *(int4*)(Bs + buf * TILE_M * LDSS + row * LDSS + seg * 8) =
                *(const int4*)(Bw + (size_t)(nBase + row) * HID + k0 + seg * 8);
        }
    };
#endif

#if HAVE_TDM
    if (wave == 0) {
        tdm_load_tile(As, Aw, HID);
        tdm_load_tile(Bs, Bw + (size_t)nBase * HID, HID);
        tdm_wait();
    }
#else
    stage(0, 0);
#endif
    __syncthreads();

    const int ksteps = HID / TILE_K;   // 128
    for (int kt = 0; kt < ksteps; ++kt) {
        int buf = kt & 1;
        if (kt + 1 < ksteps) {
            int k0 = (kt + 1) * TILE_K;
#if HAVE_TDM
            if (wave == 0) {
                tdm_load_tile(As + (buf ^ 1) * TILE_M * LDSS, Aw + k0, HID);
                tdm_load_tile(Bs + (buf ^ 1) * TILE_M * LDSS,
                              Bw + (size_t)nBase * HID + k0, HID);
            }
#else
            stage(k0, buf ^ 1);
#endif
        }
        v16bf af[2], bfg[4];
#pragma unroll
        for (int i = 0; i < 2; ++i)
            af[i] = ldsFragA(As + buf * TILE_M * LDSS + (wm * 32 + i * 16) * LDSS, lane);
#pragma unroll
        for (int j = 0; j < 4; ++j)
            bfg[j] = ldsFragB(Bs + buf * TILE_M * LDSS + (wn * 64 + j * 16) * LDSS, lane);
#pragma unroll
        for (int i = 0; i < 2; ++i)
#pragma unroll
            for (int j = 0; j < 4; ++j)
                acc[i][j] = __builtin_amdgcn_wmma_f32_16x16x32_bf16(
                    false, af[i], false, bfg[j], (short)0, acc[i][j], false, false);
#if HAVE_TDM
        if (wave == 0 && kt + 1 < ksteps) tdm_wait();
#endif
        __syncthreads();
    }

    // Epilogue: fp32 restage in LDS, two 64-column passes, coalesced b128 stores.
    const int CFS = 68;                          // 272B rows
    float* Cf = (float*)smem;                    // 128*68*4 = 34816B
    int half = lane >> 4, ln = lane & 15;
#pragma unroll
    for (int p = 0; p < 2; ++p) {
        if (wn == p) {
#pragma unroll
            for (int i = 0; i < 2; ++i)
#pragma unroll
                for (int j = 0; j < 4; ++j)
#pragma unroll
                    for (int r = 0; r < 8; ++r)
                        Cf[(wm * 32 + i * 16 + half * 8 + r) * CFS + j * 16 + ln] =
                            acc[i][j][r];
        }
        __syncthreads();
#pragma unroll
        for (int it = 0; it < 8; ++it) {
            int c = tid + it * 256;              // 2048 x 16B chunks
            int r = c >> 4, seg = c & 15;
            *(float4*)(Gout + (size_t)(rowBase + r) * DIM + nBase + p * 64 + seg * 4) =
                *(const float4*)(Cf + r * CFS + seg * 4);
        }
        __syncthreads();
    }
}

// out[t, :] = sum_k w[t,k] * Gout[rowOf[t,k], :]
__global__ void combine_kernel(const float* __restrict__ Gout, const float* __restrict__ wts,
                               const int* __restrict__ rowOf, float* __restrict__ out) {
    int t = blockIdx.x;
    int d = threadIdx.x * 4;
    float4 o = make_float4(0.f, 0.f, 0.f, 0.f);
#pragma unroll
    for (int k = 0; k < TOPK; ++k) {
        float w  = wts[t * TOPK + k];
        int  row = rowOf[t * TOPK + k];
        float4 g = *(const float4*)(Gout + (size_t)row * DIM + d);
        o.x += w * g.x; o.y += w * g.y; o.z += w * g.z; o.w += w * g.w;
    }
    *(float4*)(out + (size_t)t * DIM + d) = o;
}

// ---------------------------------------------------------------------------
extern "C" void kernel_launch(void* const* d_in, const int* in_sizes, int n_in,
                              void* d_out, int out_size, void* d_ws, size_t ws_size,
                              hipStream_t stream) {
    (void)in_sizes; (void)n_in; (void)out_size; (void)ws_size;
    const float* x        = (const float*)d_in[0];
    const float* router_w = (const float*)d_in[1];
    const float* up_w     = (const float*)d_in[2];
    const float* down_w   = (const float*)d_in[3];
    float*       out      = (float*)d_out;

    char*  ws = (char*)d_ws;
    size_t o  = 0;
    auto alloc = [&](size_t bytes) -> void* {
        o = (o + 255) & ~(size_t)255;
        void* p = ws + o;
        o += bytes;
        return p;
    };

    __bf16* xb      = (__bf16*)alloc((size_t)TOKENS * DIM * 2);
    __bf16* upT     = (__bf16*)alloc((size_t)NEXP * HID * DIM * 2);
    __bf16* downT   = (__bf16*)alloc((size_t)NEXP * DIM * HID * 2);
    float*  wts     = (float*) alloc((size_t)TOKENS * TOPK * 4);
    int*    idx     = (int*)   alloc((size_t)TOKENS * TOPK * 4);
    int*    counts  = (int*)   alloc(NEXP * 4);
    int*    fill    = (int*)   alloc(NEXP * 4);
    int*    offsets = (int*)   alloc(NEXP * 4);
    int*    tokList = (int*)   alloc((size_t)MAX_ROWS * 4);
    int*    rowOf   = (int*)   alloc((size_t)TOKENS * TOPK * 4);
    __bf16* Hbuf    = (__bf16*)alloc((size_t)MAX_ROWS * HID * 2);
    float*  Gout    = (float*) alloc((size_t)MAX_ROWS * DIM * 4);

    (void)hipMemsetAsync(counts, 0, NEXP * 4, stream);
    (void)hipMemsetAsync(fill,   0, NEXP * 4, stream);

    convert_x_bf16<<<(TOKENS * DIM / 4) / 256, 256, 0, stream>>>(x, xb);
    transpose_to_bf16<<<dim3(HID / 32, DIM / 32, NEXP), 256, 0, stream>>>(up_w,   upT,   DIM, HID);
    transpose_to_bf16<<<dim3(DIM / 32, HID / 32, NEXP), 256, 0, stream>>>(down_w, downT, HID, DIM);
    router_topk_kernel<<<TOKENS / 8, 256, 0, stream>>>(x, router_w, wts, idx, counts);
    scan_kernel<<<1, 32, 0, stream>>>(counts, offsets);
    scatter_kernel<<<TOKENS / 256, 256, 0, stream>>>(idx, offsets, fill, tokList, rowOf);
    gemm_up_kernel<<<dim3(HID / TILE_N, TOKENS / TILE_M, NEXP), 256, 0, stream>>>(
        xb, upT, tokList, counts, offsets, Hbuf);
    gemm_down_kernel<<<dim3(DIM / TILE_N, TOKENS / TILE_M, NEXP), 256, 0, stream>>>(
        Hbuf, downT, counts, offsets, Gout);
    combine_kernel<<<TOKENS, DIM / 4, 0, stream>>>(Gout, wts, rowOf, out);
}
